// RNN_Many2Many_26018911879256
// MI455X (gfx1250) — matrix operand order repaired and checked by reading, not compiled, not run on hardware
//
#include <hip/hip_runtime.h>
#include <hip/hip_bf16.h>

#define HID    128
#define TSTEPS 1024
#define BATCH  256
#define ROWS   16            // batch rows per block (one WMMA M-tile)
#define NWAVES 9             // 8 compute waves + 1 FC wave
#define THREADS (NWAVES * 32)

typedef _Float16 v8h  __attribute__((ext_vector_type(8)));
typedef _Float16 v16h __attribute__((ext_vector_type(16)));
typedef float    v8f  __attribute__((ext_vector_type(8)));

// Hardware v_tanh_f32 (confirmed emitted on gfx1250); branch-free fallback.
__device__ __forceinline__ float fast_tanh(float x) {
#if __has_builtin(__builtin_amdgcn_tanhf)
    return __builtin_amdgcn_tanhf(x);
#elif __has_builtin(__builtin_amdgcn_tanh_f32)
    return __builtin_amdgcn_tanh_f32(x);
#else
    float ax = __builtin_fabsf(x);
    float e  = __builtin_amdgcn_exp2f(ax * 2.885390081777927f);
    float r  = 1.0f - 2.0f * __builtin_amdgcn_rcpf(e + 1.0f);
    return __builtin_copysignf(r, x);
#endif
}

// 128-wide K reduction: single 4-deep WMMA accumulation chain.
// D->C accumulation is not a listed WMMA->WMMA hazard, so the chain
// pipelines at full rate; one WMMA->VALU hazard bubble total.
__device__ __forceinline__ v8f wmma_k128(
    const _Float16* __restrict__ ap, int kAoff, const v16h bm[4])
{
    v8f acc = {};
    #pragma unroll
    for (int c = 0; c < 4; ++c) {
        v8h lo = *(const v8h*)(ap + c * 32 + kAoff);
        v8h hi = *(const v8h*)(ap + c * 32 + kAoff + 16);
        v16h a = __builtin_shufflevector(lo, hi,
            0,1,2,3,4,5,6,7,8,9,10,11,12,13,14,15);
        acc = __builtin_amdgcn_wmma_f32_16x16x32_f16(
            false, a, false, bm[c], (short)0, acc, false, false);
    }
    return acc;
}

// One persistent block per 16-row batch tile. Per step:
//   waves 0..7 : h_t(:, 16w..16w+15) = tanh(xp_t + h_{t-1} @ Whh^T)
//   wave  8    : logits_{t-1} = h_{t-1} @ Wfc^T + b_fc (one step behind)
__global__ __launch_bounds__(THREADS) void rnn_persistent(
    const float* __restrict__ x,      // [B][T]
    const float* __restrict__ W_ih,   // [128]
    const float* __restrict__ W_hh,   // [128][128]
    const float* __restrict__ b_ih,   // [128]
    const float* __restrict__ b_hh,   // [128]
    const float* __restrict__ W_fc,   // [2][128]
    const float* __restrict__ b_fc,   // [2]
    float* __restrict__ out)          // [B][T][2]
{
    __shared__ _Float16 hbuf[2][ROWS * HID];   // ping-pong h tiles (f16)
    __shared__ _Float16 xT[TSTEPS * ROWS];     // x transposed: [t][row]

    const int tid   = threadIdx.x;
    const int wave  = tid >> 5;
    const int lane  = tid & 31;
    const int col   = lane & 15;               // C/D: N index
    const int rbase = (lane < 16) ? 0 : 8;     // C/D: M = rbase + vgpr
    const int kAoff = (lane < 16) ? 0 : 8;     // A: K sub-offset within 32-chunk
    const int kBoff = (lane < 16) ? 0 : 8;     // B: K sub-offset within 32-chunk
    const int bbase = blockIdx.x * ROWS;

    // ---- stage x tile into LDS, transposed, as f16 (coalesced along t) ----
    for (int idx = tid; idx < ROWS * TSTEPS; idx += THREADS) {
        int r = idx >> 10;                 // 0..15
        int t = idx & (TSTEPS - 1);
        xT[t * ROWS + r] = (_Float16)x[(size_t)(bbase + r) * TSTEPS + t];
    }
    // h_{-1} = 0
    for (int idx = tid; idx < ROWS * HID; idx += THREADS)
        hbuf[0][idx] = (_Float16)0.f;

    // ---- pin B panel in VGPRs: Whh^T slice (waves 0-7) or Wfc^T (wave 8) ----
    const float* brow = nullptr;
    if (wave < 8)      brow = W_hh + (size_t)(wave * 16 + col) * HID;  // B[k][n]=Whh[n][k]
    else if (col < 2)  brow = W_fc + (size_t)col * HID;                // B[k][n]=Wfc[n][k]
    v16h bm[4];
    #pragma unroll
    for (int c = 0; c < 4; ++c) {
        v16h v = {};
        if (brow) {
            const float* bp = brow + c * 32 + kBoff;
            #pragma unroll
            for (int i = 0; i < 16; ++i) v[i] = (_Float16)bp[i < 8 ? i : i + 8];
        }
        bm[c] = v;
    }

    // per-lane scalars
    float w_ih_c = 0.f, bias_c = 0.f;
    if (wave < 8) {
        int gcol = wave * 16 + col;
        w_ih_c = W_ih[gcol];
        bias_c = b_ih[gcol] + b_hh[gcol];
    } else if (col < 2) {
        bias_c = b_fc[col];
    }

    __syncthreads();

    for (int t = 0; t <= TSTEPS; ++t) {
        const _Float16* hsrc = hbuf[t & 1];            // holds h_{t-1}
        if (wave < 8) {
            if (t < TSTEPS) {
                const _Float16* ap = hsrc + (lane & 15) * HID;  // A row = lane&15
                v8f acc = wmma_k128(ap, kAoff, bm);
                v8h xv = *(const v8h*)(xT + t * ROWS + rbase);
                _Float16* hdst = hbuf[(t + 1) & 1] + rbase * HID + (wave * 16 + col);
                #pragma unroll
                for (int i = 0; i < 8; ++i) {
                    float v = acc[i] + (float)xv[i] * w_ih_c + bias_c;
                    hdst[i * HID] = (_Float16)fast_tanh(v);
                }
            }
        } else {
            if (t > 0) {   // FC on h_{t-1}
                const _Float16* ap = hsrc + (lane & 15) * HID;
                v8f acc = wmma_k128(ap, kAoff, bm);
                if (col < 2) {
                    #pragma unroll
                    for (int i = 0; i < 8; ++i) {
                        size_t b = (size_t)(bbase + rbase + i);
                        out[(b * TSTEPS + (size_t)(t - 1)) * 2 + col] = acc[i] + bias_c;
                    }
                }
            }
        }
        __syncthreads();
    }
}

extern "C" void kernel_launch(void* const* d_in, const int* in_sizes, int n_in,
                              void* d_out, int out_size, void* d_ws, size_t ws_size,
                              hipStream_t stream) {
    const float* x    = (const float*)d_in[0];
    const float* W_ih = (const float*)d_in[1];
    const float* W_hh = (const float*)d_in[2];
    const float* b_ih = (const float*)d_in[3];
    const float* b_hh = (const float*)d_in[4];
    const float* W_fc = (const float*)d_in[5];
    const float* b_fc = (const float*)d_in[6];
    float* out = (float*)d_out;

    rnn_persistent<<<BATCH / ROWS, THREADS, 0, stream>>>(
        x, W_ih, W_hh, b_ih, b_hh, W_fc, b_fc, out);
}
